// AttentionHead_10754598110053
// MI455X (gfx1250) — matrix operand order repaired and checked by reading, not compiled, not run on hardware
//
#include <hip/hip_runtime.h>
#include <hip/hip_bf16.h>

// Problem constants (match reference)
#define BATCH  4
#define SEQ    2048
#define DMODEL 1024
#define DH     64
#define ROWS   (BATCH * SEQ)   // 8192
#define WPB    4               // waves per block in attention kernel

typedef __attribute__((ext_vector_type(8)))  __bf16 v8bf;
typedef __attribute__((ext_vector_type(16))) __bf16 v16bf;
typedef __attribute__((ext_vector_type(8)))  float  v8f;
typedef __attribute__((ext_vector_type(4)))  float  v4f;
typedef __attribute__((__vector_size__(16))) int    i32x4;   // matches builtin

#define LOG2E 1.4426950408889634f

// ---- CDNA5 async global->LDS (builtin confirmed present on this toolchain) --
#if __has_builtin(__builtin_amdgcn_global_load_async_to_lds_b128)
#define ASYNC_LDS 1
#pragma message("CDNA5: using __builtin_amdgcn_global_load_async_to_lds_b128")
#else
#define ASYNC_LDS 0
#pragma message("CDNA5: async-to-LDS builtin NOT available, sync fallback")
#endif
#if __has_builtin(__builtin_amdgcn_s_wait_asynccnt)
#define HAVE_WAIT_ASYNC 1
#else
#define HAVE_WAIT_ASYNC 0
#endif
#if __has_builtin(__builtin_amdgcn_permlane16)
#define HAVE_PERMLANE16 1
#pragma message("CDNA5: using v_permlane16_b32 for row reductions")
#else
#define HAVE_PERMLANE16 0
#endif

// Copy 16 aligned bytes global -> LDS (async DMA on CDNA5, sync fallback).
__device__ __forceinline__ void cp16_g2l(const __bf16* g, __bf16* l) {
#if ASYNC_LDS
    i32x4* gg = (i32x4*)(void*)g;     // strip const, retype to int4
    i32x4* ll = (i32x4*)(void*)l;
    __builtin_amdgcn_global_load_async_to_lds_b128(
        (__attribute__((address_space(1))) i32x4*)gg,
        (__attribute__((address_space(3))) i32x4*)ll,
        0, 0);
#else
    *(v8bf*)l = *(const v8bf*)g;
#endif
}

__device__ __forceinline__ void wait_async_then_barrier() {
#if ASYNC_LDS
#if HAVE_WAIT_ASYNC
    __builtin_amdgcn_s_wait_asynccnt(0);
#else
    asm volatile("s_wait_asynccnt 0x0" ::: "memory");
#endif
#endif
    __syncthreads();
}

// xor-shuffle within each 16-lane row on the VALU pipe (v_permlane16_b32),
// keeping the LDS pipe free for fragments + async DMA.
template <int M>
__device__ __forceinline__ float rowshfl_xor(float v) {
#if HAVE_PERMLANE16
    constexpr unsigned s0 = (M == 1) ? 0x67452301u :
                            (M == 2) ? 0x54761032u :
                            (M == 4) ? 0x32107654u : 0xFEDCBA98u;
    constexpr unsigned s1 = (M == 1) ? 0xEFCDAB89u :
                            (M == 2) ? 0xDCFE98BAu :
                            (M == 4) ? 0xBA98FEDCu : 0x76543210u;
    int i = __float_as_int(v);
    i = __builtin_amdgcn_permlane16(i, i, s0, s1, false, false);
    return __int_as_float(i);
#else
    return __shfl_xor(v, M, 32);
#endif
}

__device__ __forceinline__ float rowmax16(float v) {
    v = fmaxf(v, rowshfl_xor<1>(v));
    v = fmaxf(v, rowshfl_xor<2>(v));
    v = fmaxf(v, rowshfl_xor<4>(v));
    v = fmaxf(v, rowshfl_xor<8>(v));
    return v;
}
__device__ __forceinline__ float rowsum16(float v) {
    v += rowshfl_xor<1>(v);
    v += rowshfl_xor<2>(v);
    v += rowshfl_xor<4>(v);
    v += rowshfl_xor<8>(v);
    return v;
}

// Concatenate two 8-wide bf16 vectors (two b128 loads) into a 16-wide fragment.
__device__ __forceinline__ v16bf cat8(v8bf lo, v8bf hi) {
    return __builtin_shufflevector(lo, hi, 0, 1, 2, 3, 4, 5, 6, 7,
                                           8, 9, 10, 11, 12, 13, 14, 15);
}

// ---------------------------------------------------------------------------
// Kernel 0: Wt[n][k] = bf16(W[k][n])   (64 x 1024 per matrix)
// ---------------------------------------------------------------------------
__global__ __launch_bounds__(256)
void wt_kernel(const float* __restrict__ W, __bf16* __restrict__ Wt) {
    const int idx = blockIdx.x * 256 + threadIdx.x;   // n*1024 + k
    const int n = idx >> 10;
    const int k = idx & 1023;
    Wt[idx] = (__bf16)W[(size_t)k * DH + n];
}

// ---------------------------------------------------------------------------
// Kernel 1: out = bf16( scale * (x @ W + Bias) ), one wave per 16x64 tile.
// ---------------------------------------------------------------------------
template <int TRANSPOSED>
__global__ __launch_bounds__(32)
void qkv_proj_kernel(const float* __restrict__ x,
                     const __bf16* __restrict__ Wt,
                     const float* __restrict__ Bias,
                     __bf16* __restrict__ out,
                     float scale)
{
    const int t    = blockIdx.x;          // 16-row tile index, 0..511
    const int lane = threadIdx.x & 31;
    const int hi   = lane >> 4;
    const int l15  = lane & 15;

    v8f c[4];
    #pragma unroll
    for (int nt = 0; nt < 4; ++nt)
        #pragma unroll
        for (int r = 0; r < 8; ++r)
            c[nt][r] = 0.0f;

    const float* xrow = x + (size_t)(t * 16 + l15) * DMODEL;

    for (int kb = 0; kb < DMODEL / 32; ++kb) {
        // A fragment: two contiguous 8-float runs per lane, converted to bf16.
        const float* xr = xrow + kb * 32;
        v4f x0 = *(const v4f*)(xr + 8 * hi);
        v4f x1 = *(const v4f*)(xr + 8 * hi + 4);
        v4f x2 = *(const v4f*)(xr + 16 + 8 * hi);
        v4f x3 = *(const v4f*)(xr + 16 + 8 * hi + 4);
        v16bf a;
        #pragma unroll
        for (int e = 0; e < 4; ++e) {
            a[e]      = (__bf16)x0[e];
            a[4 + e]  = (__bf16)x1[e];
            a[8 + e]  = (__bf16)x2[e];
            a[12 + e] = (__bf16)x3[e];
        }
        #pragma unroll
        for (int nt = 0; nt < 4; ++nt) {
            const __bf16* wr = Wt + (size_t)(nt * 16 + l15) * DMODEL
                             + kb * 32 + 8 * hi;
            v16bf bm = cat8(*(const v8bf*)wr, *(const v8bf*)(wr + 16));
            c[nt] = __builtin_amdgcn_wmma_f32_16x16x32_bf16(
                false, a, false, bm, (short)0, c[nt], false, false);
        }
    }

    #pragma unroll
    for (int nt = 0; nt < 4; ++nt) {
        #pragma unroll
        for (int r = 0; r < 8; ++r) {
            const int row = t * 16 + r + 8 * hi;      // C-layout: elem r -> M
            const int b   = row >> 11;                // row / SEQ
            const int s   = row & (SEQ - 1);
            const int n   = nt * 16 + l15;
            float v = (c[nt][r] + Bias[(size_t)s * DH + n]) * scale;
            if (TRANSPOSED)
                out[((size_t)b * DH + n) * SEQ + s] = (__bf16)v;  // Vt[b][d][s]
            else
                out[(size_t)row * DH + n] = (__bf16)v;            // [row][d]
        }
    }
}

// ---------------------------------------------------------------------------
// Kernel 2: flash attention, 64-key tiles, double-buffered async LDS staging.
// Block = 4 waves = 4 consecutive 16-query tiles of one batch.
// Per tile: 8 score WMMAs (2 d-chunks x 4 key quarters), one softmax pass
// (4+4 permlane16 reductions per row), 8 PV WMMAs.
// ---------------------------------------------------------------------------
__global__ __launch_bounds__(WPB * 32)
void attn_kernel(const __bf16* __restrict__ Q,
                 const __bf16* __restrict__ K,
                 const __bf16* __restrict__ Vt,
                 float* __restrict__ out)
{
    __shared__ __bf16 Kt_lds[2][64 * 64];     // [buf][key][dim]  2x 8 KB
    __shared__ __bf16 Vt_lds[2][64 * 64];     // [buf][dim][key]  2x 8 KB
    __shared__ __bf16 Pb[WPB][16 * 64];       // P bounce            8 KB

    const int tid  = threadIdx.x;
    const int wid  = tid >> 5;
    const int lane = tid & 31;
    const int hi   = lane >> 4;
    const int l15  = lane & 15;

    const int b     = blockIdx.x / (SEQ / 64);       // batch
    const int grp   = blockIdx.x % (SEQ / 64);       // 64-query group
    const int qbase = grp * 64 + wid * 16;

    const __bf16* Qb  = Q  + (size_t)b * SEQ * DH;
    const __bf16* Kb  = K  + (size_t)b * SEQ * DH;
    const __bf16* Vtb = Vt + (size_t)b * DH * SEQ;

    // Q A-fragments (head-dim chunks [0,32) and [32,64)): 2x b128 each.
    const __bf16* qr = Qb + (size_t)(qbase + l15) * DH;
    v16bf aq0 = cat8(*(const v8bf*)(qr + 8 * hi),
                     *(const v8bf*)(qr + 16 + 8 * hi));
    v16bf aq1 = cat8(*(const v8bf*)(qr + 32 + 8 * hi),
                     *(const v8bf*)(qr + 48 + 8 * hi));

    float mrun[8], lrun[8];
    v8f o[4];
    #pragma unroll
    for (int r = 0; r < 8; ++r) { mrun[r] = -__builtin_inff(); lrun[r] = 0.0f; }
    #pragma unroll
    for (int t = 0; t < 4; ++t)
        #pragma unroll
        for (int r = 0; r < 8; ++r) o[t][r] = 0.0f;

    // Stage one 64-key tile (K: 8 KB contiguous; Vt: 64 rows x 128 B).
    auto stage = [&](int buf, int jb) {
        #pragma unroll
        for (int p = 0; p < 4; ++p) {
            const int c = tid + p * 128;              // 0..511, 16 B chunks
            cp16_g2l(Kb + (size_t)jb * DH + c * 8, &Kt_lds[buf][c * 8]);
            const int dim = c >> 3, part = c & 7;
            cp16_g2l(Vtb + (size_t)dim * SEQ + jb + part * 8,
                     &Vt_lds[buf][dim * 64 + part * 8]);
        }
    };

    stage(0, 0);   // prologue: fill buffer 0

    for (int j = 0; j < SEQ / 64; ++j) {
        const int cur = j & 1;

        // Current buffer's DMA has landed for every wave -> safe to read.
        wait_async_then_barrier();

        // Kick off next tile's DMA into the other buffer; its last readers
        // finished before the end-of-iteration barrier one iteration ago.
        if (j + 1 < SEQ / 64)
            stage(cur ^ 1, (j + 1) * 64);

        // ---- S = Q * K^T over 4 key quarters (8 WMMAs) ---------------------
        v8f s[4];
        #pragma unroll
        for (int q = 0; q < 4; ++q) {
            const __bf16* kr = &Kt_lds[cur][(q * 16 + l15) * 64];
            v16bf bk0 = cat8(*(const v8bf*)(kr + 8 * hi),
                             *(const v8bf*)(kr + 16 + 8 * hi));
            v16bf bk1 = cat8(*(const v8bf*)(kr + 32 + 8 * hi),
                             *(const v8bf*)(kr + 48 + 8 * hi));
            v8f acc;
            #pragma unroll
            for (int r = 0; r < 8; ++r) acc[r] = 0.0f;
            acc = __builtin_amdgcn_wmma_f32_16x16x32_bf16(
                false, aq0, false, bk0, (short)0, acc, false, false);
            acc = __builtin_amdgcn_wmma_f32_16x16x32_bf16(
                false, aq1, false, bk1, (short)0, acc, false, false);
            s[q] = acc;
        }

        // ---- online softmax: each row lives in one 16-lane half ------------
        #pragma unroll
        for (int r = 0; r < 8; ++r) {
            const float v0 = s[0][r], v1 = s[1][r], v2 = s[2][r], v3 = s[3][r];
            float mx = fmaxf(fmaxf(v0, v1), fmaxf(v2, v3));
            mx = rowmax16(mx);
            const float mnew = fmaxf(mrun[r], mx);
            const float p0 = exp2f((v0 - mnew) * LOG2E);
            const float p1 = exp2f((v1 - mnew) * LOG2E);
            const float p2 = exp2f((v2 - mnew) * LOG2E);
            const float p3 = exp2f((v3 - mnew) * LOG2E);
            float rs = (p0 + p1) + (p2 + p3);
            rs = rowsum16(rs);
            const float alpha = exp2f((mrun[r] - mnew) * LOG2E);
            lrun[r] = lrun[r] * alpha + rs;
            mrun[r] = mnew;
            #pragma unroll
            for (int t = 0; t < 4; ++t) o[t][r] *= alpha;
            const int prow = r + 8 * hi;
            Pb[wid][prow * 64 + l15]      = (__bf16)p0;
            Pb[wid][prow * 64 + 16 + l15] = (__bf16)p1;
            Pb[wid][prow * 64 + 32 + l15] = (__bf16)p2;
            Pb[wid][prow * 64 + 48 + l15] = (__bf16)p3;
        }

        // ---- P back as two A-frags (per-wave LDS, in-order) ----------------
        const __bf16* pr = &Pb[wid][l15 * 64];
        v16bf ap0 = cat8(*(const v8bf*)(pr + 8 * hi),
                         *(const v8bf*)(pr + 16 + 8 * hi));
        v16bf ap1 = cat8(*(const v8bf*)(pr + 32 + 8 * hi),
                         *(const v8bf*)(pr + 48 + 8 * hi));

        // ---- O += P * V: 8 WMMAs, B-frags from transposed V tile -----------
        #pragma unroll
        for (int t = 0; t < 4; ++t) {
            const __bf16* vr = &Vt_lds[cur][(t * 16 + l15) * 64];
            v16bf bv0 = cat8(*(const v8bf*)(vr + 8 * hi),
                             *(const v8bf*)(vr + 16 + 8 * hi));
            v16bf bv1 = cat8(*(const v8bf*)(vr + 32 + 8 * hi),
                             *(const v8bf*)(vr + 48 + 8 * hi));
            o[t] = __builtin_amdgcn_wmma_f32_16x16x32_bf16(
                false, ap0, false, bv0, (short)0, o[t], false, false);
            o[t] = __builtin_amdgcn_wmma_f32_16x16x32_bf16(
                false, ap1, false, bv1, (short)0, o[t], false, false);
        }

        __syncthreads();   // all waves done reading tile j
    }

    // Normalize and store fp32 output [B, S, 64].
    #pragma unroll
    for (int r = 0; r < 8; ++r) {
        const float inv = 1.0f / lrun[r];
        const int row = qbase + r + 8 * hi;
        #pragma unroll
        for (int t = 0; t < 4; ++t)
            out[((size_t)b * SEQ + row) * DH + t * 16 + l15] = o[t][r] * inv;
    }
}

// ---------------------------------------------------------------------------
// Launcher. Inputs (setup_inputs order): x, Wq, Bq, Wk, Bk, Wv, Bv (all fp32).
// Output: fp32 [B, S, 64].  Workspace (~3.4 MB).
// ---------------------------------------------------------------------------
extern "C" void kernel_launch(void* const* d_in, const int* in_sizes, int n_in,
                              void* d_out, int out_size, void* d_ws, size_t ws_size,
                              hipStream_t stream) {
    const float* x  = (const float*)d_in[0];
    const float* Wq = (const float*)d_in[1];
    const float* Bq = (const float*)d_in[2];
    const float* Wk = (const float*)d_in[3];
    const float* Bk = (const float*)d_in[4];
    const float* Wv = (const float*)d_in[5];
    const float* Bv = (const float*)d_in[6];
    float* out = (float*)d_out;

    __bf16* qws  = (__bf16*)d_ws;                       // 1 MB
    __bf16* kws  = qws + (size_t)ROWS * DH;             // 1 MB
    __bf16* vtws = kws + (size_t)ROWS * DH;             // 1 MB ([B][64][S])
    __bf16* wqt  = vtws + (size_t)ROWS * DH;            // 128 KB each
    __bf16* wkt  = wqt + (size_t)DH * DMODEL;
    __bf16* wvt  = wkt + (size_t)DH * DMODEL;

    // Weight transpose + bf16 convert: Wt[n][k].
    dim3 g0((DH * DMODEL) / 256), b0(256);
    wt_kernel<<<g0, b0, 0, stream>>>(Wq, wqt);
    wt_kernel<<<g0, b0, 0, stream>>>(Wk, wkt);
    wt_kernel<<<g0, b0, 0, stream>>>(Wv, wvt);

    // Projections. Q pre-scaled by 1/sqrt(64) = 0.125 (exact power of two).
    dim3 g1(ROWS / 16), b1(32);
    qkv_proj_kernel<0><<<g1, b1, 0, stream>>>(x, wqt, Bq, qws,  0.125f);
    qkv_proj_kernel<0><<<g1, b1, 0, stream>>>(x, wkt, Bk, kws,  1.0f);
    qkv_proj_kernel<1><<<g1, b1, 0, stream>>>(x, wvt, Bv, vtws, 1.0f);

    // Attention: 128 blocks x 4 waves, each wave one 16-query tile.
    dim3 g2(BATCH * (SEQ / 64)), b2(WPB * 32);
    attn_kernel<<<g2, b2, 0, stream>>>(qws, kws, vtws, out);
}
